// GRUD_16123307229743
// MI455X (gfx1250) — matrix operand order repaired and checked by reading, not compiled, not run on hardware
//
#include <hip/hip_runtime.h>

// ---------------------------------------------------------------------------
// GRU-D for MI455X (gfx1250, wave32, WMMA bf16 16x16x32, f32 accumulate)
//
// B=512, T=128, D=128, H=256, C=512.
// Phase 1 (parallel): elementwise x-imputation; pack weights to bf16.
// Phase 2 (parallel): WMMA GEMMs for delta_h and the [x,m]-column partials of
//                     the z/r/h~ projections (biases folded in), f32 outputs.
// Phase 3 (sequential): persistent kernel, 32 blocks x 8 waves, 2 tiles/wave
//                     covering the [512 x 256] state; 2 grid barriers/step.
// Phase 4 (parallel): binary head sigmoid(hs @ fc_w.T + fc_b).
// ---------------------------------------------------------------------------

typedef __bf16 bf16_t;
typedef __attribute__((ext_vector_type(16))) __bf16 v16bf;
typedef __attribute__((ext_vector_type(8)))  __bf16 v8bf;
typedef __attribute__((ext_vector_type(8)))  float  v8f;

static constexpr int Bc = 512;
static constexpr int Tc = 128;
static constexpr int Dc = 128;
static constexpr int Hc = 256;
static constexpr int Cc = 512;   // D + H + D
static constexpr int Mrows = Tc * Bc;          // 65536 rows for parallel GEMMs
static constexpr int REC_BLOCKS = 32;          // persistent recurrent grid

union FragU { v16bf v; struct { v8bf lo; v8bf hi; } s; };

// A-matrix 16x32 bf16 fragment (ISA 7.12.2): lane<16 -> M=lane, K={0..7,16..23};
// lane>=16 -> M=lane-16, K={8..15,24..31}. Two contiguous 16B loads per lane.
__device__ __forceinline__ v16bf load_a_frag(const bf16_t* __restrict__ base,
                                             int ldk, int m0, int k0, int lane) {
  const int half = lane >> 4;
  const int m    = lane & 15;
  const bf16_t* row = base + (size_t)(m0 + m) * ldk + k0 + half * 8;
  FragU f;
  f.s.lo = *(const v8bf*)(row);
  f.s.hi = *(const v8bf*)(row + 16);
  return f.v;
}

// B-matrix 32x16 bf16 fragment: lane -> N=lane&15, K = (lane>>4)*8 + {0..7,16..23}.
// Weights stored row-major [N][K] (i.e. W so that out = A @ W^T): contiguous.
__device__ __forceinline__ v16bf load_b_frag(const bf16_t* __restrict__ w,
                                             int ldk, int n0, int k0, int lane) {
  const int half = lane >> 4;
  const int n    = lane & 15;
  const bf16_t* row = w + (size_t)(n0 + n) * ldk + k0 + half * 8;
  FragU f;
  f.s.lo = *(const v8bf*)(row);
  f.s.hi = *(const v8bf*)(row + 16);
  return f.v;
}

__device__ __forceinline__ v8f wmma_bf16(v16bf a, v16bf b, v8f c) {
  return __builtin_amdgcn_wmma_f32_16x16x32_bf16(false, a, false, b,
                                                 (short)0, c, false, false);
}

__device__ __forceinline__ float sigmoidf_(float x) {
  return 1.0f / (1.0f + __expf(-x));
}

// Grid-wide sense barrier (all REC_BLOCKS blocks must be co-resident).
__device__ __forceinline__ void grid_sync(unsigned* cnt, volatile unsigned* gen,
                                          unsigned nblk) {
  __syncthreads();
  if (threadIdx.x == 0) {
    __threadfence();
    unsigned g = *gen;
    if (atomicAdd(cnt, 1u) == nblk - 1u) {
      *cnt = 0u;
      __threadfence();
      atomicAdd((unsigned*)gen, 1u);
    } else {
      while (*gen == g) { __builtin_amdgcn_s_sleep(1); }
    }
    __threadfence();
  }
  __syncthreads();
}

// ---------------------------------------------------------------------------
// K1: pack weights to bf16 layouts. W*_xm = columns [0:128)+[384:512) of the
// [H,512] weights (x and m parts, K=256); W*_h = columns [128:384) (h part).
// Also resets the grid barrier state.
// ---------------------------------------------------------------------------
__global__ void k1_pack(const float* __restrict__ zl_w, const float* __restrict__ rl_w,
                        const float* __restrict__ hl_w, const float* __restrict__ gh_w,
                        bf16_t* Wz_xm, bf16_t* Wr_xm, bf16_t* Wh_xm,
                        bf16_t* Wz_h,  bf16_t* Wr_h,  bf16_t* Wh_h,
                        bf16_t* Wgh, unsigned* bar) {
  const int tid = blockIdx.x * blockDim.x + threadIdx.x;
  const int nt  = gridDim.x * blockDim.x;
  for (int i = tid; i < Hc * 256; i += nt) {
    const int j = i >> 8, k = i & 255;
    const int src = (k < Dc) ? k : (Dc + Hc + (k - Dc));   // x part | m part
    Wz_xm[i] = (bf16_t)zl_w[j * Cc + src];
    Wr_xm[i] = (bf16_t)rl_w[j * Cc + src];
    Wh_xm[i] = (bf16_t)hl_w[j * Cc + src];
    Wz_h[i]  = (bf16_t)zl_w[j * Cc + Dc + k];
    Wr_h[i]  = (bf16_t)rl_w[j * Cc + Dc + k];
    Wh_h[i]  = (bf16_t)hl_w[j * Cc + Dc + k];
  }
  for (int i = tid; i < Hc * Dc; i += nt) Wgh[i] = (bf16_t)gh_w[i];
  if (tid == 0) { bar[0] = 0u; bar[1] = 0u; }
}

// ---------------------------------------------------------------------------
// K2: elementwise decay-imputation of x; emit bf16 [x|m] rows (K=256) and
// bf16 Delta rows (K=128). Row index = t*B + b.
// ---------------------------------------------------------------------------
__global__ void k2_impute(const float* __restrict__ input,   // [B,4,T,D]
                          const float* __restrict__ X_mean,  // [1,T,D]
                          const float* __restrict__ gx_w,    // [D,D] (diag used)
                          const float* __restrict__ gx_b,    // [D]
                          bf16_t* __restrict__ xm,           // [T*B, 256]
                          bf16_t* __restrict__ db) {         // [T*B, 128]
  const int idx = blockIdx.x * blockDim.x + threadIdx.x;     // over T*B*D
  if (idx >= Mrows * Dc) return;
  const int d   = idx & (Dc - 1);
  const int row = idx >> 7;            // t*B + b
  const int t   = row >> 9;            // /512
  const int b   = row & (Bc - 1);
  const size_t cstride = (size_t)Tc * Dc;
  const size_t base = (size_t)b * 4 * cstride + (size_t)t * Dc + d;
  const float x   = input[base + 0 * cstride];
  const float xl  = input[base + 1 * cstride];
  const float m   = input[base + 2 * cstride];
  const float del = input[base + 3 * cstride];
  const float xme = X_mean[t * Dc + d];
  float v = del * gx_w[d * Dc + d] + gx_b[d];
  v = v > 0.f ? v : 0.f;
  const float dx = __expf(-v);
  const float xx = m * x + (1.f - m) * (dx * xl + (1.f - dx) * xme);
  xm[(size_t)row * 256 + d]       = (bf16_t)xx;
  xm[(size_t)row * 256 + Dc + d]  = (bf16_t)m;
  db[(size_t)row * Dc + d]        = (bf16_t)del;
}

// ---------------------------------------------------------------------------
// K3: time-parallel WMMA GEMMs.
//  pre_g[t,b,:] = [x|m] @ Wg_xm^T + bias   (g in {z,r,h}, K=256)
//  dh[t,b,:]    = exp(-relu(Delta @ gh_w^T + gh_b))  (K=128)
// One 16x16 tile per wave; 8 waves/block; 65536 tiles per output.
// ---------------------------------------------------------------------------
__global__ void k3_pregemm(const bf16_t* __restrict__ xm, const bf16_t* __restrict__ db,
                           const bf16_t* __restrict__ Wz, const bf16_t* __restrict__ Wr,
                           const bf16_t* __restrict__ Wh, const bf16_t* __restrict__ Wg,
                           const float* __restrict__ zb, const float* __restrict__ rb,
                           const float* __restrict__ hb, const float* __restrict__ gb,
                           float* __restrict__ pre_z, float* __restrict__ pre_r,
                           float* __restrict__ pre_h, float* __restrict__ dh) {
  const int lane = threadIdx.x & 31;
  const int wave = threadIdx.x >> 5;
  const int task = blockIdx.x * 8 + wave;       // 65536 tiles
  const int ct = task & 15;                     // col tile (H/16)
  const int rt = task >> 4;                     // row tile (M/16)
  const int m0 = rt * 16;
  const int n0 = ct * 16;

  v16bf a_xm[8], a_db[4];
#pragma unroll
  for (int k = 0; k < 8; k++) a_xm[k] = load_a_frag(xm, 256, m0, k * 32, lane);
#pragma unroll
  for (int k = 0; k < 4; k++) a_db[k] = load_a_frag(db, Dc, m0, k * 32, lane);

  const int half = lane >> 4, n = lane & 15;
  const int gcol = n0 + n;

  const bf16_t* Ws[3] = { Wz, Wr, Wh };
  const float*  bs[3] = { zb, rb, hb };
  float*        os[3] = { pre_z, pre_r, pre_h };
#pragma unroll
  for (int g = 0; g < 3; g++) {
    v8f c = {};
#pragma unroll
    for (int k = 0; k < 8; k++)
      c = wmma_bf16(a_xm[k], load_b_frag(Ws[g], 256, n0, k * 32, lane), c);
    const float bias = bs[g][gcol];
#pragma unroll
    for (int r = 0; r < 8; r++) {
      const int gm = m0 + r + 8 * half;
      os[g][(size_t)gm * Hc + gcol] = c[r] + bias;
    }
  }
  {
    v8f c = {};
#pragma unroll
    for (int k = 0; k < 4; k++)
      c = wmma_bf16(a_db[k], load_b_frag(Wg, Dc, n0, k * 32, lane), c);
    const float bias = gb[gcol];
#pragma unroll
    for (int r = 0; r < 8; r++) {
      const int gm = m0 + r + 8 * half;
      float v = c[r] + bias;
      v = v > 0.f ? v : 0.f;
      dh[(size_t)gm * Hc + gcol] = __expf(-v);
    }
  }
}

// ---------------------------------------------------------------------------
// K4: persistent recurrent kernel. 32 blocks x 8 waves; each wave owns two
// adjacent 16x16 tiles (same 16 rows of B, 32 cols of H) of the [512,256]
// state. Per step:
//   phase B: z = sig(pre_z + hm @ Wz_h^T); r = sig(pre_r + hm @ Wr_h^T);
//            write rh = r*hm (bf16).                       [barrier]
//   phase C: h~ = tanh(pre_h + rh @ Wh_h^T);
//            h_new = (1-z)*hm + z*h~; hs[t] = h_new;
//            hm <- dh[t+1]*h_new (fused decay, bf16 to hm_buf). [barrier]
// ---------------------------------------------------------------------------
__global__ void k4_recurrent(const float* __restrict__ pre_z, const float* __restrict__ pre_r,
                             const float* __restrict__ pre_h, const float* __restrict__ dh,
                             const bf16_t* __restrict__ Wz_h, const bf16_t* __restrict__ Wr_h,
                             const bf16_t* __restrict__ Wh_h,
                             bf16_t* __restrict__ hm_buf,   // [512,256] bf16
                             bf16_t* __restrict__ rh_buf,   // [512,256] bf16
                             float* __restrict__ hs,        // [T*B, 256] f32
                             unsigned* bar) {
  unsigned* cnt = bar;
  volatile unsigned* gen = bar + 1;

  const int lane = threadIdx.x & 31;
  const int wave = threadIdx.x >> 5;
  const int tid0 = (blockIdx.x * 8 + wave) * 2;   // first of two tiles
  const int rt = tid0 >> 4;                       // row tile (B/16): 0..31
  const int c0 = tid0 & 15;                       // col tiles c0, c0+1
  const int m0 = rt * 16;
  const int half = lane >> 4, n = lane & 15;

  int brow[8];
#pragma unroll
  for (int r = 0; r < 8; r++) brow[r] = m0 + r + 8 * half;
  const int n0c[2]  = { c0 * 16, (c0 + 1) * 16 };
  const int jcol[2] = { n0c[0] + n, n0c[1] + n };

  float hm[2][8];   // decayed hidden state, owned tile elements (C layout)
  float zv[2][8];
#pragma unroll
  for (int c = 0; c < 2; c++)
#pragma unroll
    for (int r = 0; r < 8; r++) {
      hm[c][r] = 0.f;
      hm_buf[(size_t)brow[r] * Hc + jcol[c]] = (bf16_t)0.f;
    }
  grid_sync(cnt, gen, REC_BLOCKS);

  for (int t = 0; t < Tc; t++) {
    const size_t step = (size_t)t * Bc;
    // ---- phase B: gates ----
    v16bf a[8];
#pragma unroll
    for (int k = 0; k < 8; k++) a[k] = load_a_frag(hm_buf, Hc, m0, k * 32, lane);
#pragma unroll
    for (int c = 0; c < 2; c++) {
      v8f cz, cr;
#pragma unroll
      for (int r = 0; r < 8; r++) {
        const size_t o = (step + brow[r]) * Hc + jcol[c];
        cz[r] = pre_z[o];
        cr[r] = pre_r[o];
      }
#pragma unroll
      for (int k = 0; k < 8; k++) {
        cz = wmma_bf16(a[k], load_b_frag(Wz_h, Hc, n0c[c], k * 32, lane), cz);
        cr = wmma_bf16(a[k], load_b_frag(Wr_h, Hc, n0c[c], k * 32, lane), cr);
      }
#pragma unroll
      for (int r = 0; r < 8; r++) {
        const float z  = sigmoidf_(cz[r]);
        const float rr = sigmoidf_(cr[r]);
        zv[c][r] = z;
        rh_buf[(size_t)brow[r] * Hc + jcol[c]] = (bf16_t)(rr * hm[c][r]);
      }
    }
    grid_sync(cnt, gen, REC_BLOCKS);

    // ---- phase C: candidate + update + fused decay for next step ----
#pragma unroll
    for (int k = 0; k < 8; k++) a[k] = load_a_frag(rh_buf, Hc, m0, k * 32, lane);
#pragma unroll
    for (int c = 0; c < 2; c++) {
      v8f ch;
#pragma unroll
      for (int r = 0; r < 8; r++)
        ch[r] = pre_h[(step + brow[r]) * Hc + jcol[c]];
#pragma unroll
      for (int k = 0; k < 8; k++)
        ch = wmma_bf16(a[k], load_b_frag(Wh_h, Hc, n0c[c], k * 32, lane), ch);
#pragma unroll
      for (int r = 0; r < 8; r++) {
        const float ht = tanhf(ch[r]);
        const float z  = zv[c][r];
        const float hn = (1.f - z) * hm[c][r] + z * ht;
        hs[(step + brow[r]) * Hc + jcol[c]] = hn;
        float hmn = 0.f;
        if (t + 1 < Tc)
          hmn = dh[((size_t)(t + 1) * Bc + brow[r]) * Hc + jcol[c]] * hn;
        hm[c][r] = hmn;
        hm_buf[(size_t)brow[r] * Hc + jcol[c]] = (bf16_t)hmn;
      }
    }
    grid_sync(cnt, gen, REC_BLOCKS);
  }
}

// ---------------------------------------------------------------------------
// K5: head — out[b,t] = sigmoid(hs[t,b,:] . fc_w + fc_b). One wave per row,
// deterministic in-wave reduction (no FP atomics).
// ---------------------------------------------------------------------------
__global__ void k5_head(const float* __restrict__ hs, const float* __restrict__ fc_w,
                        const float* __restrict__ fc_b, float* __restrict__ out) {
  const int lane = threadIdx.x & 31;
  const int wave = threadIdx.x >> 5;
  const int row  = blockIdx.x * 8 + wave;     // t*B + b, rows = T*B
  const float* hp = hs + (size_t)row * Hc;
  float s = 0.f;
#pragma unroll
  for (int i = 0; i < 8; i++) s += hp[lane + 32 * i] * fc_w[lane + 32 * i];
#pragma unroll
  for (int off = 16; off > 0; off >>= 1) s += __shfl_xor(s, off, 32);
  if (lane == 0) {
    const int t = row >> 9, b = row & (Bc - 1);
    out[b * Tc + t] = sigmoidf_(s + fc_b[0]);
  }
}

// ---------------------------------------------------------------------------
extern "C" void kernel_launch(void* const* d_in, const int* in_sizes, int n_in,
                              void* d_out, int out_size, void* d_ws, size_t ws_size,
                              hipStream_t stream) {
  const float* input  = (const float*)d_in[0];
  const float* X_mean = (const float*)d_in[1];
  const float* zl_w = (const float*)d_in[2];  const float* zl_b = (const float*)d_in[3];
  const float* rl_w = (const float*)d_in[4];  const float* rl_b = (const float*)d_in[5];
  const float* hl_w = (const float*)d_in[6];  const float* hl_b = (const float*)d_in[7];
  const float* gx_w = (const float*)d_in[8];  const float* gx_b = (const float*)d_in[9];
  const float* gh_w = (const float*)d_in[10]; const float* gh_b = (const float*)d_in[11];
  const float* fc_w = (const float*)d_in[12]; const float* fc_b = (const float*)d_in[13];
  float* out = (float*)d_out;

  size_t off = 0;
  auto take = [&](size_t bytes) -> void* {
    void* r = (char*)d_ws + off;
    off += (bytes + 255) & ~(size_t)255;
    return r;
  };
  bf16_t* Wz_xm = (bf16_t*)take((size_t)Hc * 256 * 2);
  bf16_t* Wr_xm = (bf16_t*)take((size_t)Hc * 256 * 2);
  bf16_t* Wh_xm = (bf16_t*)take((size_t)Hc * 256 * 2);
  bf16_t* Wz_h  = (bf16_t*)take((size_t)Hc * 256 * 2);
  bf16_t* Wr_h  = (bf16_t*)take((size_t)Hc * 256 * 2);
  bf16_t* Wh_h  = (bf16_t*)take((size_t)Hc * 256 * 2);
  bf16_t* Wgh   = (bf16_t*)take((size_t)Hc * Dc * 2);
  bf16_t* xm    = (bf16_t*)take((size_t)Mrows * 256 * 2);
  bf16_t* db    = (bf16_t*)take((size_t)Mrows * Dc * 2);
  float*  pre_z = (float*)take((size_t)Mrows * Hc * 4);
  float*  pre_r = (float*)take((size_t)Mrows * Hc * 4);
  float*  pre_h = (float*)take((size_t)Mrows * Hc * 4);
  float*  dh    = (float*)take((size_t)Mrows * Hc * 4);
  float*  hs    = (float*)take((size_t)Mrows * Hc * 4);
  bf16_t* hm_buf = (bf16_t*)take((size_t)Bc * Hc * 2);
  bf16_t* rh_buf = (bf16_t*)take((size_t)Bc * Hc * 2);
  unsigned* bar  = (unsigned*)take(256);

  k1_pack<<<64, 256, 0, stream>>>(zl_w, rl_w, hl_w, gh_w,
                                  Wz_xm, Wr_xm, Wh_xm, Wz_h, Wr_h, Wh_h, Wgh, bar);

  const int n_ew = Mrows * Dc;
  k2_impute<<<(n_ew + 255) / 256, 256, 0, stream>>>(input, X_mean, gx_w, gx_b, xm, db);

  // 65536 tiles per output matrix, 8 waves/block -> 8192 blocks
  k3_pregemm<<<(Mrows / 16) * (Hc / 16) / 8, 256, 0, stream>>>(
      xm, db, Wz_xm, Wr_xm, Wh_xm, Wgh, zl_b, rl_b, hl_b, gh_b,
      pre_z, pre_r, pre_h, dh);

  k4_recurrent<<<REC_BLOCKS, 256, 0, stream>>>(pre_z, pre_r, pre_h, dh,
                                               Wz_h, Wr_h, Wh_h,
                                               hm_buf, rh_buf, hs, bar);

  k5_head<<<Mrows / 8, 256, 0, stream>>>(hs, fc_w, fc_b, out);
}
